// SCLoss_21277267984580
// MI455X (gfx1250) — matrix operand order repaired and checked by reading, not compiled, not run on hardware
//
#include <hip/hip_runtime.h>

// ---------------------------------------------------------------------------
// SCLoss pipeline for gfx1250 (MI455X), wave32.
//
//   K0 convert_w1    : W1 -> W1h + W1l (split-f16, exactness trick)
//   K1 row_rinv      : rinv[i] = 1/max(||x_i||, 1e-12)
//   K2 gemm1         : Y1 = (x @ W1^T) * rinv[row]
//                      split-f16 WMMA 16x16x32 (xh*Wh + xl*Wh + xh*Wl),
//                      B staged in LDS, double-buffered async-to-LDS copies
//   K3 colstats_part : partial col sum/sumsq (64 blocks x 128 rows)
//   K4 colstats_final: BN scale/shift (fused gamma/beta)
//   K5 gemm2         : H = relu(scale1*Y1+shift1) @ W2^T + b2 (f32 WMMA 16x16x4)
//   K6 bn2_norm      : Z = l2norm_rows(scale2*H + shift2)
//   K7 loss_part     : per-group S = X1g @ X2g^T (f32 WMMA); dist = sqrt(2-2S)
//   K8 loss_final    : combine 64 group partials -> d_out[0]
// ---------------------------------------------------------------------------

typedef float     v2f  __attribute__((ext_vector_type(2)));
typedef float     v4f  __attribute__((ext_vector_type(4)));
typedef float     v8f  __attribute__((ext_vector_type(8)));
typedef int       v4i  __attribute__((ext_vector_type(4)));
typedef _Float16  v4h  __attribute__((ext_vector_type(4)));
typedef _Float16  v8h  __attribute__((ext_vector_type(8)));
typedef _Float16  v16h __attribute__((ext_vector_type(16)));

#define BROWS     8192
#define DIN       2048
#define DH        256
#define NUMHALF   4096
#define KGRP      64
#define WEIGHT_W  0.55f
#define BN_EPS    1e-5f
#define NORM_EPS  1e-12f

#define KCHUNK    32
#define NCHUNK    (DIN / KCHUNK)   // 64
#define G1_COLS   128              // columns per block in gemm1

// ------------------------- CDNA5 async-to-LDS path --------------------------
#if defined(__has_builtin)
#if __has_builtin(__builtin_amdgcn_global_load_async_to_lds_b128)
#define HAVE_ASYNC_LDS 1
#endif
#endif
#ifndef HAVE_ASYNC_LDS
#define HAVE_ASYNC_LDS 0
#endif

#if HAVE_ASYNC_LDS
// Param types (from clang diagnostic): (AS1 v4i*, AS3 v4i*, Ii offset, Ii cpol)
#define ASYNC_B128(gsrc, ldst)                                                 \
  __builtin_amdgcn_global_load_async_to_lds_b128(                              \
      (__attribute__((address_space(1))) v4i*)(gsrc),                          \
      (__attribute__((address_space(3))) v4i*)(ldst), 0, 0)
#if __has_builtin(__builtin_amdgcn_s_wait_asynccnt)
#define WAIT_ASYNC(n) __builtin_amdgcn_s_wait_asynccnt(n)
#else
#define WAIT_ASYNC(n) asm volatile("s_wait_asynccnt %0" ::"i"(n) : "memory")
#endif
#else
#define WAIT_ASYNC(n) ((void)0)
#endif

// ------------------------- WMMA helpers -------------------------------------
static __device__ __forceinline__ v8f wmma_f32(v2f a, v2f b, v8f c) {
  // D = A(16x4 f32) x B(4x16 f32) + C ; verified lowering round 1
  return __builtin_amdgcn_wmma_f32_16x16x4_f32(false, a, false, b, (short)0, c,
                                               false, false);
}
static __device__ __forceinline__ v8f wmma_f16(v16h a, v16h b, v8f c) {
  // D = A(16x32 f16) x B(32x16 f16) + C(f32)
  return __builtin_amdgcn_wmma_f32_16x16x32_f16(false, a, false, b, (short)0, c,
                                                false, false);
}

// ---------------- K0: split W1 into f16 hi/lo parts -------------------------
__global__ void convert_w1_kernel(const float* __restrict__ W1,
                                  _Float16* __restrict__ Wh,
                                  _Float16* __restrict__ Wl) {
  const int i = (blockIdx.x * 256 + threadIdx.x) * 4;
  const v4f w = *(const v4f*)(W1 + i);
  v4h h, l;
#pragma unroll
  for (int q = 0; q < 4; ++q) {
    _Float16 hv = (_Float16)w[q];
    h[q] = hv;
    l[q] = (_Float16)(w[q] - (float)hv);
  }
  *(v4h*)(Wh + i) = h;
  *(v4h*)(Wl + i) = l;
}

// ---------------- K1: per-row inverse L2 norm of x --------------------------
__global__ void row_rinv_kernel(const float* __restrict__ x,
                                float* __restrict__ rinv) {
  __shared__ float red[256];
  const int row = blockIdx.x;
  const int t   = threadIdx.x;
  const float* xr = x + (size_t)row * DIN;
  float s = 0.f;
#pragma unroll
  for (int i = 0; i < DIN / 256; ++i) {
    float v = xr[t + i * 256];
    s += v * v;
  }
  red[t] = s;
  __syncthreads();
  for (int off = 128; off > 0; off >>= 1) {
    if (t < off) red[t] += red[t + off];
    __syncthreads();
  }
  if (t == 0) rinv[row] = 1.0f / fmaxf(sqrtf(red[0]), NORM_EPS);
}

// ---------------- K2: Y1 = (x @ W1^T) * rinv  (split-f16 WMMA) --------------
// grid (8192/128, 256/128), block 256 (8 waves). Wave: 16 rows x 128 cols.
// B (W1h/W1l k-chunks) staged in LDS, double-buffered, async when available.
__global__ void __launch_bounds__(256)
gemm1_kernel(const float* __restrict__ x,
             const _Float16* __restrict__ W1h,
             const _Float16* __restrict__ W1l,
             const float* __restrict__ rinv,
             float* __restrict__ Y) {
  __shared__ __align__(32) _Float16 ldsBh[2][G1_COLS * KCHUNK];  // 8KB each
  __shared__ __align__(32) _Float16 ldsBl[2][G1_COLS * KCHUNK];

  const int tid  = threadIdx.x;
  const int wave = tid >> 5;
  const int lane = tid & 31;
  const int l16  = lane & 15;
  const int kh   = lane >> 4;
  const int row0 = blockIdx.x * 128 + wave * 16;
  const int col0 = blockIdx.y * G1_COLS;

  // staging role: thread -> (column, 16-half slice of the 32-k row)
  const int ccol  = tid >> 1;        // 0..127
  const int chalf = (tid & 1) * 16;  // halves offset within row
  const _Float16* gH = W1h + (size_t)(col0 + ccol) * DIN + chalf;
  const _Float16* gL = W1l + (size_t)(col0 + ccol) * DIN + chalf;
  const int ldst = ccol * KCHUNK + chalf;

  auto issue_chunk = [&](int j) {
    const int b  = j & 1;
    const int kb = j * KCHUNK;
#if HAVE_ASYNC_LDS
    ASYNC_B128(gH + kb,     &ldsBh[b][ldst]);
    ASYNC_B128(gH + kb + 8, &ldsBh[b][ldst + 8]);
    ASYNC_B128(gL + kb,     &ldsBl[b][ldst]);
    ASYNC_B128(gL + kb + 8, &ldsBl[b][ldst + 8]);
#else
    *(v8h*)&ldsBh[b][ldst]     = *(const v8h*)(gH + kb);
    *(v8h*)&ldsBh[b][ldst + 8] = *(const v8h*)(gH + kb + 8);
    *(v8h*)&ldsBl[b][ldst]     = *(const v8h*)(gL + kb);
    *(v8h*)&ldsBl[b][ldst + 8] = *(const v8h*)(gL + kb + 8);
#endif
  };

  v8f acc[8] = {};
  const float* arow = x + (size_t)(row0 + l16) * DIN;

  auto compute_chunk = [&](int j) {
    const int b  = j & 1;
    const int kb = j * KCHUNK;
    // A fragment: 16 f32 -> hi/lo f16 in registers (no extra HBM traffic).
    // 16-bit A layout: elems 0..15 <-> K kb+16*kh.. (contiguous, matches B).
    const float* ap = arow + kb + 16 * kh;
    const v4f a0 = *(const v4f*)(ap);
    const v4f a1 = *(const v4f*)(ap + 4);
    const v4f a2 = *(const v4f*)(ap + 8);
    const v4f a3 = *(const v4f*)(ap + 12);
    float av[16];
#pragma unroll
    for (int q = 0; q < 4; ++q) {
      av[q] = a0[q]; av[4 + q] = a1[q]; av[8 + q] = a2[q]; av[12 + q] = a3[q];
    }
    v16h ah, al;
#pragma unroll
    for (int q = 0; q < 16; ++q) {
      _Float16 h = (_Float16)av[q];
      ah[q] = h;
      al[q] = (_Float16)(av[q] - (float)h);
    }
    // B fragments from LDS: lane n=col, elems = 16 consecutive K at 16*kh.
#pragma unroll
    for (int t = 0; t < 8; ++t) {
      const int nl = t * 16 + l16;
      v16h bh = *(const v16h*)&ldsBh[b][nl * KCHUNK + 16 * kh];
      v16h bl = *(const v16h*)&ldsBl[b][nl * KCHUNK + 16 * kh];
      acc[t] = wmma_f16(ah, bh, acc[t]);
      acc[t] = wmma_f16(al, bh, acc[t]);
      acc[t] = wmma_f16(ah, bl, acc[t]);
    }
  };

  issue_chunk(0);
  for (int i = 0; i < NCHUNK - 1; ++i) {
    issue_chunk(i + 1);
    WAIT_ASYNC(4);        // chunk i complete (in-order), chunk i+1 in flight
    __syncthreads();      // cross-wave visibility of chunk i in LDS
    compute_chunk(i);
    __syncthreads();      // all reads of buf (i&1) done before it is re-filled
  }
  WAIT_ASYNC(0);
  __syncthreads();
  compute_chunk(NCHUNK - 1);

  float rv[8];
#pragma unroll
  for (int r = 0; r < 8; ++r) rv[r] = rinv[row0 + r + 8 * kh];
#pragma unroll
  for (int t = 0; t < 8; ++t) {
#pragma unroll
    for (int r = 0; r < 8; ++r) {
      Y[(size_t)(row0 + r + 8 * kh) * DH + col0 + t * 16 + l16] =
          acc[t][r] * rv[r];
    }
  }
}

// ---------------- K3: partial column sums / sumsq ---------------------------
__global__ void colstats_part_kernel(const float* __restrict__ P,
                                     float* __restrict__ ps,
                                     float* __restrict__ pq) {
  const int col  = threadIdx.x;
  const int base = blockIdx.x * 128;
  float s = 0.f, q = 0.f;
  for (int r = 0; r < 128; ++r) {
    float v = P[(size_t)(base + r) * DH + col];
    s += v;
    q += v * v;
  }
  ps[blockIdx.x * DH + col] = s;
  pq[blockIdx.x * DH + col] = q;
}

// ---------------- K4: fold BN stats + gamma/beta into scale/shift -----------
__global__ void colstats_final_kernel(const float* __restrict__ ps,
                                      const float* __restrict__ pq,
                                      const float* __restrict__ gamma,
                                      const float* __restrict__ beta,
                                      float* __restrict__ scale,
                                      float* __restrict__ shift) {
  const int col = threadIdx.x;
  float s = 0.f, q = 0.f;
  for (int b = 0; b < 64; ++b) {
    s += ps[b * DH + col];
    q += pq[b * DH + col];
  }
  const float mean = s * (1.0f / (float)BROWS);
  const float var  = q * (1.0f / (float)BROWS) - mean * mean;
  const float istd = rsqrtf(var + BN_EPS);
  const float sc   = gamma[col] * istd;
  scale[col] = sc;
  shift[col] = beta[col] - mean * sc;
}

// ---------------- K5: H = relu(scale1*Y1+shift1) @ W2^T + b2  (f32 WMMA) ----
__global__ void gemm2_kernel(const float* __restrict__ Y,
                             const float* __restrict__ W2,
                             const float* __restrict__ scale1,
                             const float* __restrict__ shift1,
                             const float* __restrict__ b2,
                             float* __restrict__ H) {
  const int tid  = threadIdx.x;
  const int wave = tid >> 5;
  const int lane = tid & 31;
  const int l16  = lane & 15;
  const int kh   = lane >> 4;
  const int row0 = blockIdx.x * 128 + wave * 16;
  const int col0 = blockIdx.y * 64;

  const float* aptr = Y  + (size_t)(row0 + l16) * DH + 2 * kh;
  const float* bptr = W2 + (size_t)(col0 + l16) * DH + 2 * kh;

  v8f c[4] = {};
  for (int kb = 0; kb < DH; kb += 4) {
    v2f sc = *(const v2f*)(scale1 + kb + 2 * kh);
    v2f sh = *(const v2f*)(shift1 + kb + 2 * kh);
    v2f a  = *(const v2f*)(aptr + kb);
    a.x = fmaxf(a.x * sc.x + sh.x, 0.0f);  // fused BN1 + ReLU on A-load
    a.y = fmaxf(a.y * sc.y + sh.y, 0.0f);
#pragma unroll
    for (int t = 0; t < 4; ++t) {
      v2f b = *(const v2f*)(bptr + (size_t)t * 16 * DH + kb);
      c[t] = wmma_f32(a, b, c[t]);
    }
  }

#pragma unroll
  for (int t = 0; t < 4; ++t) {
    const float bias = b2[col0 + t * 16 + l16];
#pragma unroll
    for (int r = 0; r < 8; ++r) {
      H[(size_t)(row0 + r + 8 * kh) * DH + col0 + t * 16 + l16] =
          c[t][r] + bias;
    }
  }
}

// ---------------- K6: Z = l2norm_rows(scale2*H + shift2) --------------------
__global__ void bn2_norm_kernel(const float* __restrict__ H,
                                const float* __restrict__ scale2,
                                const float* __restrict__ shift2,
                                float* __restrict__ Z) {
  __shared__ float red[256];
  const int row = blockIdx.x;
  const int t   = threadIdx.x;
  const float v = H[(size_t)row * DH + t] * scale2[t] + shift2[t];
  red[t] = v * v;
  __syncthreads();
  for (int off = 128; off > 0; off >>= 1) {
    if (t < off) red[t] += red[t + off];
    __syncthreads();
  }
  const float rn = 1.0f / fmaxf(sqrtf(red[0]), NORM_EPS);
  Z[(size_t)row * DH + t] = v * rn;
}

// ---------------- K7: per-group pairwise distances via f32 WMMA -------------
__global__ void loss_part_kernel(const float* __restrict__ Z,
                                 float* __restrict__ part_all,
                                 float* __restrict__ part_diag) {
  __shared__ float redA[256];
  __shared__ float redD[256];
  const int g    = blockIdx.x;
  const int tid  = threadIdx.x;
  const int wave = tid >> 5;
  const int lane = tid & 31;
  const int l16  = lane & 15;
  const int kh   = lane >> 4;

  const int ti0  = wave * 2;
  const int rowA = (ti0 >> 2) * 16;
  const int colB = (ti0 & 3) * 16;

  const float* aptr  = Z + (size_t)(g * KGRP + rowA + l16) * DH + 2 * kh;
  const float* bptr0 = Z + (size_t)(NUMHALF + g * KGRP + colB + l16) * DH + 2 * kh;
  const float* bptr1 = bptr0 + (size_t)16 * DH;

  v8f c0 = {}, c1 = {};
  for (int kb = 0; kb < DH; kb += 4) {
    v2f a  = *(const v2f*)(aptr + kb);
    v2f b0 = *(const v2f*)(bptr0 + kb);
    v2f b1 = *(const v2f*)(bptr1 + kb);
    c0 = wmma_f32(a, b0, c0);
    c1 = wmma_f32(a, b1, c1);
  }

  float sa = 0.f, sd = 0.f;
#pragma unroll
  for (int r = 0; r < 8; ++r) {
    const int i  = rowA + r + 8 * kh;
    const int j0 = colB + l16;
    const int j1 = colB + 16 + l16;
    float d0 = sqrtf(fmaxf(2.0f - 2.0f * c0[r], 0.0f));
    float d1 = sqrtf(fmaxf(2.0f - 2.0f * c1[r], 0.0f));
    sa += d0 + d1;
    if (i == j0) sd += d0;
    if (i == j1) sd += d1;
  }
  redA[tid] = sa;
  redD[tid] = sd;
  __syncthreads();
  for (int off = 128; off > 0; off >>= 1) {
    if (tid < off) {
      redA[tid] += redA[tid + off];
      redD[tid] += redD[tid + off];
    }
    __syncthreads();
  }
  if (tid == 0) {
    part_all[g]  = redA[0];
    part_diag[g] = redD[0];
  }
}

// ---------------- K8: final weighted combine --------------------------------
__global__ void loss_final_kernel(const float* __restrict__ part_all,
                                  const float* __restrict__ part_diag,
                                  float* __restrict__ out) {
  __shared__ float rA[64];
  __shared__ float rD[64];
  const int t = threadIdx.x;
  rA[t] = part_all[t];
  rD[t] = part_diag[t];
  __syncthreads();
  for (int off = 32; off > 0; off >>= 1) {
    if (t < off) {
      rA[t] += rA[t + off];
      rD[t] += rD[t + off];
    }
    __syncthreads();
  }
  if (t == 0) {
    const float off_w = (1.0f - WEIGHT_W) / (float)(KGRP - 1);
    out[0] = (off_w * rA[0] + (WEIGHT_W - off_w) * rD[0]) * (1.0f / (float)NUMHALF);
  }
}

// ---------------------------------------------------------------------------
extern "C" void kernel_launch(void* const* d_in, const int* in_sizes, int n_in,
                              void* d_out, int out_size, void* d_ws, size_t ws_size,
                              hipStream_t stream) {
  (void)in_sizes; (void)n_in; (void)out_size; (void)ws_size;

  const float* x   = (const float*)d_in[0];
  const float* W1  = (const float*)d_in[1];
  const float* g1  = (const float*)d_in[2];
  const float* be1 = (const float*)d_in[3];
  const float* W2  = (const float*)d_in[4];
  const float* b2  = (const float*)d_in[5];
  const float* g2  = (const float*)d_in[6];
  const float* be2 = (const float*)d_in[7];
  // d_in[8] = label1 (int32): k=64 fixed by construction, unused.

  float* ws = (float*)d_ws;
  float* rinv      = ws;                        // 8192
  float* Y1        = rinv + BROWS;              // 8192*256
  float* H         = Y1 + (size_t)BROWS * DH;   // 8192*256
  float* Z         = H  + (size_t)BROWS * DH;   // 8192*256
  float* ps        = Z  + (size_t)BROWS * DH;   // 64*256
  float* pq        = ps + 64 * DH;              // 64*256
  float* scale1    = pq + 64 * DH;              // 256
  float* shift1    = scale1 + DH;               // 256
  float* scale2    = shift1 + DH;               // 256
  float* shift2    = scale2 + DH;               // 256
  float* part_all  = shift2 + DH;               // 64
  float* part_diag = part_all + 64;             // 64
  _Float16* W1h    = (_Float16*)(part_diag + 64);   // 256*2048 halves
  _Float16* W1l    = W1h + (size_t)DH * DIN;        // 256*2048 halves

  // K0: split W1 -> f16 hi/lo (524288 elems / 4 per thread / 256 per block)
  convert_w1_kernel<<<dim3(512), dim3(256), 0, stream>>>(W1, W1h, W1l);

  // K1: per-row inverse norms of x
  row_rinv_kernel<<<dim3(BROWS), dim3(256), 0, stream>>>(x, rinv);

  // K2: Y1 = (x @ W1^T) * rinv   (split-f16 WMMA, async LDS staging)
  gemm1_kernel<<<dim3(BROWS / 128, DH / G1_COLS), dim3(256), 0, stream>>>(
      x, W1h, W1l, rinv, Y1);

  // K3/K4: BN1 stats -> scale1/shift1
  colstats_part_kernel<<<dim3(64), dim3(256), 0, stream>>>(Y1, ps, pq);
  colstats_final_kernel<<<dim3(1), dim3(256), 0, stream>>>(ps, pq, g1, be1,
                                                           scale1, shift1);

  // K5: H = relu(bn1(Y1)) @ W2^T + b2
  gemm2_kernel<<<dim3(BROWS / 128, DH / 64), dim3(256), 0, stream>>>(
      Y1, W2, scale1, shift1, b2, H);

  // K3/K4: BN2 stats -> scale2/shift2
  colstats_part_kernel<<<dim3(64), dim3(256), 0, stream>>>(H, ps, pq);
  colstats_final_kernel<<<dim3(1), dim3(256), 0, stream>>>(ps, pq, g2, be2,
                                                           scale2, shift2);

  // K6: Z = l2norm(bn2(H))
  bn2_norm_kernel<<<dim3(BROWS), dim3(256), 0, stream>>>(H, scale2, shift2, Z);

  // K7/K8: grouped pairwise distances -> loss
  loss_part_kernel<<<dim3(64), dim3(256), 0, stream>>>(Z, part_all, part_diag);
  loss_final_kernel<<<dim3(1), dim3(64), 0, stream>>>(part_all, part_diag,
                                                      (float*)d_out);
}
